// TransformerAttention_1005022347760
// MI455X (gfx1250) — matrix operand order repaired and checked by reading, not compiled, not run on hardware
//
#include <hip/hip_runtime.h>
#include <hip/hip_bf16.h>

// ---------------------------------------------------------------------------
// MI455X (gfx1250) attention, bf16 WMMA 16x16x32 + fp32 accum.
// Compute-bound (~137 GFLOP vs ~8us HBM) -> maximize v_wmma density.
// GEMMs: 128x128 macro tile, 8 waves (32x64 each, 8 WMMAs/K-step),
// double-buffered LDS fed by global_load_async_to_lds_b128 (ASYNCcnt DMA).
// ---------------------------------------------------------------------------

typedef __attribute__((ext_vector_type(16))) __bf16 v16bf;
typedef __attribute__((ext_vector_type(8)))  __bf16 v8bf;
typedef __attribute__((ext_vector_type(8)))  float  v8f;
typedef int v4i_t __attribute__((vector_size(16)));   // matches builtin proto

constexpr int Bb  = 4;
constexpr int Ss  = 2048;
constexpr int Hh  = 1024;
constexpr int NHh = 16;
constexpr int HDd = 64;
constexpr int Mtot = Bb * Ss;                    // 8192
constexpr size_t OUTSZ = (size_t)Bb * Ss * Hh;   // 8388608

#ifndef __has_builtin
#define __has_builtin(x) 0
#endif
#if __has_builtin(__builtin_amdgcn_global_load_async_to_lds_b128)
#define HAVE_ASYNC_LDS 1
#else
#define HAVE_ASYNC_LDS 0
#endif

static __device__ __forceinline__ v8f wmma_bf16(v16bf a, v16bf b, v8f c) {
  return __builtin_amdgcn_wmma_f32_16x16x32_bf16(
      false, a, false, b, (short)0, c, false, false);
}

// 16-byte global->LDS copy: async DMA (ASYNCcnt) when available.
static __device__ __forceinline__ void copy16(const __bf16* g, __bf16* l) {
#if HAVE_ASYNC_LDS
  __builtin_amdgcn_global_load_async_to_lds_b128(
      (__attribute__((address_space(1))) v4i_t*)g,
      (__attribute__((address_space(3))) v4i_t*)l, 0, 0);
#else
  *(uint4*)l = *(const uint4*)g;
#endif
}

static __device__ __forceinline__ void wait_async() {
#if HAVE_ASYNC_LDS
#if __has_builtin(__builtin_amdgcn_s_wait_asynccnt)
  __builtin_amdgcn_s_wait_asynccnt(0);
#else
  asm volatile("s_wait_asynccnt 0x0" ::: "memory");
#endif
#endif
}

// ---------------------------------------------------------------------------
// fp32 -> bf16 elementwise (n % 8 == 0)
// ---------------------------------------------------------------------------
__global__ __launch_bounds__(256) void cvt_bf16_kernel(
    const float* __restrict__ src, __bf16* __restrict__ dst, int n) {
  const int i = (blockIdx.x * 256 + threadIdx.x) * 8;
  if (i < n) {
    float4 a = *(const float4*)(src + i);
    float4 b = *(const float4*)(src + i + 4);
    v8bf o;
    o[0] = (__bf16)a.x; o[1] = (__bf16)a.y; o[2] = (__bf16)a.z; o[3] = (__bf16)a.w;
    o[4] = (__bf16)b.x; o[5] = (__bf16)b.y; o[6] = (__bf16)b.z; o[7] = (__bf16)b.w;
    *(v8bf*)(dst + i) = o;
  }
}

// ---------------------------------------------------------------------------
// GEMM: Yf[M,N] = Xb[M,K](bf16) @ Wb[N,K](bf16)^T + bias[N](f32)
//   headLayout==1: write as [B,NH,S,HD]; ==0: flat [M,N].
// 256 thr = 8 waves; macro 128x128; wave 32x64; double-buffered async LDS.
// ---------------------------------------------------------------------------
__global__ __launch_bounds__(256) void gemm_bf16_kernel(
    const __bf16* __restrict__ X, const __bf16* __restrict__ W,
    const float* __restrict__ bias, float* __restrict__ Yf, int headLayout) {
  __shared__ alignas(32) __bf16 Xl[2][128 * 32];
  __shared__ alignas(32) __bf16 Wl[2][128 * 32];

  const int tid  = threadIdx.x;
  const int lane = tid & 31;
  const int wv   = tid >> 5;          // 0..7
  const int wm   = (wv & 3) * 32;     // wave M offset (4 rows of waves)
  const int wn   = (wv >> 2) * 64;    // wave N offset (2 cols of waves)
  const int hi   = lane >> 4;
  const int ln   = lane & 15;
  const int m0   = blockIdx.x * 128;
  const int n0   = blockIdx.y * 128;

  v8f acc[2][4];
#pragma unroll
  for (int mb = 0; mb < 2; ++mb)
#pragma unroll
    for (int nb = 0; nb < 4; ++nb) acc[mb][nb] = (v8f){};

  // staging: each tile is 128 rows x 32 cols bf16 = 512 x 16B chunks,
  // 256 threads x 2 chunks. chunk c: row = c>>2, col-offset = (c&3)*8.
  auto stage = [&](int buf, int k0) {
#pragma unroll
    for (int cc = 0; cc < 2; ++cc) {
      const int c   = tid + cc * 256;
      const int row = c >> 2;
      const int off = (c & 3) * 8;
      copy16(X + (size_t)(m0 + row) * Hh + k0 + off, &Xl[buf][row * 32 + off]);
      copy16(W + (size_t)(n0 + row) * Hh + k0 + off, &Wl[buf][row * 32 + off]);
    }
  };

  stage(0, 0);
  wait_async();
  __syncthreads();

  for (int k0 = 0; k0 < Hh; k0 += 32) {
    const int cur = (k0 >> 5) & 1;
    const int nxt = cur ^ 1;
    if (k0 + 32 < Hh) stage(nxt, k0 + 32);   // overlap copy with WMMAs

    // A fragments (16x32 bf16): lanes 0-15 K 0-15; lanes 16-31 K 16-31
    v16bf afr[2];
#pragma unroll
    for (int mb = 0; mb < 2; ++mb) {
      const __bf16* xr = &Xl[cur][(wm + mb * 16 + ln) * 32];
      v8bf lo = *(const v8bf*)(xr + hi * 16);
      v8bf hg = *(const v8bf*)(xr + hi * 16 + 8);
#pragma unroll
      for (int j = 0; j < 8; ++j) { afr[mb][j] = lo[j]; afr[mb][j + 8] = hg[j]; }
    }
    // B fragments (32x16 bf16): lane = col; lanes 0-15 K 0-15, 16-31 K 16-31
#pragma unroll
    for (int nb = 0; nb < 4; ++nb) {
      v16bf bfr = *(const v16bf*)&Wl[cur][(wn + nb * 16 + ln) * 32 + hi * 16];
#pragma unroll
      for (int mb = 0; mb < 2; ++mb)
        acc[mb][nb] = wmma_bf16(afr[mb], bfr, acc[mb][nb]);
    }
    wait_async();
    __syncthreads();
  }

  // epilogue: C/D layout vgpr r -> row r (+8 upper half-wave)
#pragma unroll
  for (int mb = 0; mb < 2; ++mb)
#pragma unroll
    for (int nb = 0; nb < 4; ++nb) {
      const int gn = n0 + wn + nb * 16 + ln;
      const float bv = bias[gn];
#pragma unroll
      for (int r = 0; r < 8; ++r) {
        const int gm = m0 + wm + mb * 16 + r + hi * 8;
        const float val = acc[mb][nb][r] + bv;
        if (headLayout) {
          const int bb = gm >> 11, s = gm & (Ss - 1);
          const int hh = gn >> 6, dd = gn & (HDd - 1);
          Yf[(((size_t)(bb * NHh + hh) * Ss + s) << 6) + dd] = val;
        } else {
          Yf[(size_t)gm * Hh + gn] = val;
        }
      }
    }
}

// ---------------------------------------------------------------------------
// Flash attention: block (4 waves) per (b,h,64-query block); 64-key tiles.
// fp32 q/k/v (required outputs) -> bf16 LDS; scores WMMA; online softmax;
// P via per-wave LDS (C->A layout); PV WMMA; ctx written bf16 [B,S,H] to ws.
// ---------------------------------------------------------------------------
__global__ __launch_bounds__(128) void attn_kernel(
    const float* __restrict__ q, const float* __restrict__ k,
    const float* __restrict__ v, const int* __restrict__ amask,
    __bf16* __restrict__ ctxb) {
  __shared__ alignas(32) __bf16 Kl[64 * 64];      // [key][d]
  __shared__ alignas(32) __bf16 Vt[64 * 64];      // [d][key]
  __shared__ alignas(32) __bf16 Pl[4 * 16 * 64];  // per-wave [row][key]

  const int tid  = threadIdx.x;
  const int lane = tid & 31;
  const int wv   = tid >> 5;
  const int hi   = lane >> 4;
  const int ln   = lane & 15;
  const int bh   = blockIdx.y;
  const int b    = bh >> 4;
  const int h    = bh & (NHh - 1);
  const int s0   = blockIdx.x * 64;

  const float* qbase = q + (size_t)bh * Ss * HDd;
  const float* kbase = k + (size_t)bh * Ss * HDd;
  const float* vbase = v + (size_t)bh * Ss * HDd;

  // persistent Q fragments (two K-steps over HD=64)
  v16bf aq0, aq1;
  {
    const float* qr = qbase + (size_t)(s0 + wv * 16 + ln) * HDd;
    const int kb = hi * 16;
#pragma unroll
    for (int j = 0; j < 8; ++j) {
      aq0[j] = (__bf16)qr[kb + j];
      aq1[j] = (__bf16)qr[32 + kb + j];
    }
#pragma unroll
    for (int j = 8; j < 16; ++j) {
      aq0[j] = (__bf16)qr[kb + j];
      aq1[j] = (__bf16)qr[32 + kb + j];
    }
  }

  v8f o0 = {}, o1 = {}, o2 = {}, o3 = {};
  float mi[8], li[8];
#pragma unroll
  for (int r = 0; r < 8; ++r) { mi[r] = -1e30f; li[r] = 0.0f; }

  const int srowS = tid >> 1;
  const int segS  = (tid & 1) * 32;

  for (int key0 = 0; key0 < Ss; key0 += 64) {
    // stage K [key][d] and V transposed [d][key], fp32 -> bf16
    {
      const float* ks = kbase + (size_t)(key0 + srowS) * HDd + segS;
      __bf16* kd = &Kl[srowS * 64 + segS];
#pragma unroll
      for (int i = 0; i < 32; i += 4) {
        float4 f = *(const float4*)(ks + i);
        kd[i + 0] = (__bf16)f.x; kd[i + 1] = (__bf16)f.y;
        kd[i + 2] = (__bf16)f.z; kd[i + 3] = (__bf16)f.w;
      }
      const float* vs = vbase + (size_t)(key0 + srowS) * HDd + segS;
#pragma unroll
      for (int i = 0; i < 32; i += 4) {
        float4 f = *(const float4*)(vs + i);
        const int d = segS + i;
        Vt[(d + 0) * 64 + srowS] = (__bf16)f.x;
        Vt[(d + 1) * 64 + srowS] = (__bf16)f.y;
        Vt[(d + 2) * 64 + srowS] = (__bf16)f.z;
        Vt[(d + 3) * 64 + srowS] = (__bf16)f.w;
      }
    }
    __syncthreads();

    float madd[4];
#pragma unroll
    for (int nt = 0; nt < 4; ++nt) {
      const int mk = amask[b * Ss + key0 + nt * 16 + ln];
      madd[nt] = (1.0f - (float)mk) * -10000.0f;
    }

    // scores = Q @ K^T
    v8f sa0 = {}, sa1 = {}, sa2 = {}, sa3 = {};
#pragma unroll
    for (int ks = 0; ks < 2; ++ks) {
      v16bf aqf = ks ? aq1 : aq0;
      const int kboff = ks * 32 + hi * 16;
      v16bf kb0 = *(const v16bf*)&Kl[(0 * 16 + ln) * 64 + kboff];
      v16bf kb1 = *(const v16bf*)&Kl[(1 * 16 + ln) * 64 + kboff];
      v16bf kb2 = *(const v16bf*)&Kl[(2 * 16 + ln) * 64 + kboff];
      v16bf kb3 = *(const v16bf*)&Kl[(3 * 16 + ln) * 64 + kboff];
      sa0 = wmma_bf16(aqf, kb0, sa0);
      sa1 = wmma_bf16(aqf, kb1, sa1);
      sa2 = wmma_bf16(aqf, kb2, sa2);
      sa3 = wmma_bf16(aqf, kb3, sa3);
    }

    // online softmax, scale 1/sqrt(64)=0.125
    float p[4][8];
#pragma unroll
    for (int r = 0; r < 8; ++r) {
      p[0][r] = sa0[r] * 0.125f + madd[0];
      p[1][r] = sa1[r] * 0.125f + madd[1];
      p[2][r] = sa2[r] * 0.125f + madd[2];
      p[3][r] = sa3[r] * 0.125f + madd[3];
    }
#pragma unroll
    for (int r = 0; r < 8; ++r) {
      float rmax = fmaxf(fmaxf(p[0][r], p[1][r]), fmaxf(p[2][r], p[3][r]));
      rmax = fmaxf(rmax, __shfl_xor(rmax, 1, 32));
      rmax = fmaxf(rmax, __shfl_xor(rmax, 2, 32));
      rmax = fmaxf(rmax, __shfl_xor(rmax, 4, 32));
      rmax = fmaxf(rmax, __shfl_xor(rmax, 8, 32));
      const float mnew  = fmaxf(mi[r], rmax);
      const float alpha = __expf(mi[r] - mnew);
      mi[r] = mnew;
      li[r] *= alpha;
      o0[r] *= alpha; o1[r] *= alpha; o2[r] *= alpha; o3[r] *= alpha;
      float rsum = 0.0f;
#pragma unroll
      for (int nt = 0; nt < 4; ++nt) {
        const float e = __expf(p[nt][r] - mnew);
        p[nt][r] = e;
        rsum += e;
      }
      rsum += __shfl_xor(rsum, 1, 32);
      rsum += __shfl_xor(rsum, 2, 32);
      rsum += __shfl_xor(rsum, 4, 32);
      rsum += __shfl_xor(rsum, 8, 32);
      li[r] += rsum;
    }

    // P: C-layout -> A-layout through per-wave LDS
    __bf16* pbase = &Pl[wv * 16 * 64];
#pragma unroll
    for (int nt = 0; nt < 4; ++nt)
#pragma unroll
      for (int r = 0; r < 8; ++r)
        pbase[(r + hi * 8) * 64 + nt * 16 + ln] = (__bf16)p[nt][r];

    // O += P @ V
#pragma unroll
    for (int ks = 0; ks < 2; ++ks) {
      v16bf ap;
      {
        const __bf16* pr = pbase + ln * 64 + ks * 32 + hi * 16;
#pragma unroll
        for (int j = 0; j < 8; ++j)  ap[j] = pr[j];
#pragma unroll
        for (int j = 8; j < 16; ++j) ap[j] = pr[j];
      }
      const int kboff = ks * 32 + hi * 16;
      v16bf vb0 = *(const v16bf*)&Vt[(0 * 16 + ln) * 64 + kboff];
      v16bf vb1 = *(const v16bf*)&Vt[(1 * 16 + ln) * 64 + kboff];
      v16bf vb2 = *(const v16bf*)&Vt[(2 * 16 + ln) * 64 + kboff];
      v16bf vb3 = *(const v16bf*)&Vt[(3 * 16 + ln) * 64 + kboff];
      o0 = wmma_bf16(ap, vb0, o0);
      o1 = wmma_bf16(ap, vb1, o1);
      o2 = wmma_bf16(ap, vb2, o2);
      o3 = wmma_bf16(ap, vb3, o3);
    }
    __syncthreads();
  }

  // normalize; store ctx bf16 [B,S,H] for the output projection
#pragma unroll
  for (int dt = 0; dt < 4; ++dt) {
    v8f oo = dt == 0 ? o0 : dt == 1 ? o1 : dt == 2 ? o2 : o3;
    const int col = h * HDd + dt * 16 + ln;
#pragma unroll
    for (int r = 0; r < 8; ++r) {
      const int s = s0 + wv * 16 + r + hi * 8;
      ctxb[(size_t)(b * Ss + s) * Hh + col] = (__bf16)(oo[r] / li[r]);
    }
  }
}

// ---------------------------------------------------------------------------
// Launch: bf16 pre-convert (x, 4 weights) -> Q/K/V GEMMs -> attention ->
// output GEMM. d_out = [out | q | k | v]. ws: xb | wq | wk | wv | wo | ctxb.
// ---------------------------------------------------------------------------
extern "C" void kernel_launch(void* const* d_in, const int* in_sizes, int n_in,
                              void* d_out, int out_size, void* d_ws, size_t ws_size,
                              hipStream_t stream) {
  (void)in_sizes; (void)n_in; (void)out_size; (void)ws_size;

  const float* x    = (const float*)d_in[0];
  const int*   mask = (const int*)d_in[1];
  const float* Wq   = (const float*)d_in[2];
  const float* bq   = (const float*)d_in[3];
  const float* Wk   = (const float*)d_in[4];
  const float* bk   = (const float*)d_in[5];
  const float* Wv   = (const float*)d_in[6];
  const float* bv   = (const float*)d_in[7];
  const float* Wo   = (const float*)d_in[8];
  const float* bo   = (const float*)d_in[9];

  float* out = (float*)d_out;
  float* qo  = out + OUTSZ;
  float* ko  = out + 2 * OUTSZ;
  float* vo  = out + 3 * OUTSZ;

  const size_t NX = (size_t)Mtot * Hh;   // 8388608
  const size_t NW = (size_t)Hh * Hh;     // 1048576
  __bf16* xb   = (__bf16*)d_ws;
  __bf16* wqb  = xb + NX;
  __bf16* wkb  = wqb + NW;
  __bf16* wvb  = wkb + NW;
  __bf16* wob  = wvb + NW;
  __bf16* ctxb = wob + NW;

  cvt_bf16_kernel<<<(int)(NX / 2048), 256, 0, stream>>>(x, xb, (int)NX);
  cvt_bf16_kernel<<<(int)(NW / 2048), 256, 0, stream>>>(Wq, wqb, (int)NW);
  cvt_bf16_kernel<<<(int)(NW / 2048), 256, 0, stream>>>(Wk, wkb, (int)NW);
  cvt_bf16_kernel<<<(int)(NW / 2048), 256, 0, stream>>>(Wv, wvb, (int)NW);
  cvt_bf16_kernel<<<(int)(NW / 2048), 256, 0, stream>>>(Wo, wob, (int)NW);

  dim3 gProj(Mtot / 128, Hh / 128);   // 64 x 8
  dim3 bProj(256);
  gemm_bf16_kernel<<<gProj, bProj, 0, stream>>>(xb, wqb, bq, qo, 1);
  gemm_bf16_kernel<<<gProj, bProj, 0, stream>>>(xb, wkb, bk, ko, 1);
  gemm_bf16_kernel<<<gProj, bProj, 0, stream>>>(xb, wvb, bv, vo, 1);

  dim3 gAttn(Ss / 64, Bb * NHh);      // 32 x 64
  attn_kernel<<<gAttn, dim3(128), 0, stream>>>(qo, ko, vo, mask, ctxb);

  gemm_bf16_kernel<<<gProj, bProj, 0, stream>>>(ctxb, wob, bo, out, 0);
}
